// LRUMLP_40243843564172
// MI455X (gfx1250) — matrix operand order repaired and checked by reading, not compiled, not run on hardware
//
#include <hip/hip_runtime.h>
#include <hip/hip_bf16.h>
#include <math.h>

typedef _Float16 v16h __attribute__((ext_vector_type(16)));
typedef _Float16 v8h  __attribute__((ext_vector_type(8)));
typedef _Float16 h2   __attribute__((ext_vector_type(2)));
typedef float    v8f  __attribute__((ext_vector_type(8)));
typedef unsigned int u32x4 __attribute__((ext_vector_type(4)));
typedef int          i32x4 __attribute__((ext_vector_type(4)));
typedef int          i32x8 __attribute__((ext_vector_type(8)));

#define EDIM 256
#define HDIM 256
#define LDIM 1024
#define TDIM 2048
#define BDIM 32
#define TSDIM 64
#define ODIM 10
#define NBLK 6
#define MROWS 65536  // B*T

#if defined(__has_builtin)
#if __has_builtin(__builtin_amdgcn_tensor_load_to_lds) && \
    __has_builtin(__builtin_amdgcn_s_wait_tensorcnt)
#define USE_TDM 1
#endif
#endif

// ---------------------------------------------------------------------------
// TDM: DMA one 64x64 f16 tile (row stride = K elements) from global memory
// into LDS with rows padded 128B data + 16B pad (matches [64][72] layout).
// Issued by a single wave; EXEC ignored by hardware. Caller must
// s_wait_tensorcnt(0) on the issuing wave before the workgroup barrier.
// ---------------------------------------------------------------------------
#ifdef USE_TDM
__device__ __forceinline__ void tdm_load_tile64x64_f16(
    const _Float16* gptr, unsigned ldsOff, int K)
{
  unsigned long long ga = (unsigned long long)(size_t)gptr;
  u32x4 g0;
  g0[0] = 1u;                                            // count=1, user mode
  g0[1] = ldsOff;                                        // lds_addr
  g0[2] = (unsigned)(ga & 0xffffffffu);                  // global_addr[31:0]
  g0[3] = (unsigned)((ga >> 32) & 0x01ffffffu)           // global_addr[56:32]
        | (2u << 30);                                    // type=2 ("image")
  i32x8 g1;
  g1[0] = (1 << 16)                                      // data_size: 2 bytes
        | (1 << 20)                                      // pad_enable
        | (4 << 22)                                      // pad_interval: 32 DWORDs
        | (3 << 25);                                     // pad_amount: 4 DWORDs
  g1[1] = (int)(((unsigned)K & 0xffffu) << 16);          // tensor_dim0[15:0]
  g1[2] = (int)((((unsigned)K >> 16) & 0xffffu)          // tensor_dim0[31:16]
        | (((unsigned)MROWS & 0xffffu) << 16));          // tensor_dim1[15:0]
  g1[3] = (int)((((unsigned)MROWS >> 16) & 0xffffu)      // tensor_dim1[31:16]
        | (64u << 16));                                  // tile_dim0 = 64
  g1[4] = 64;                                            // tile_dim1 = 64 (tile_dim2=0)
  g1[5] = K;                                             // tensor_dim0_stride[31:0]
  g1[6] = 0;
  g1[7] = 0;
  i32x4 gz = {0, 0, 0, 0};
#if defined(__clang_major__) && __clang_major__ >= 23
  i32x8 gz8 = {0, 0, 0, 0, 0, 0, 0, 0};
  __builtin_amdgcn_tensor_load_to_lds(g0, g1, gz, gz, gz8, 0);
#else
  __builtin_amdgcn_tensor_load_to_lds(g0, g1, gz, gz, 0);
#endif
}
#endif

// ---------------------------------------------------------------------------
// Generic WMMA GEMM: C[M,N](f32 acc) = A[M,K](f16) * B[K,N](f16) + epilogue.
// Tile: BM=64, BN=128, BK=64 per 256-thread block (8 wave32).
// Each wave: 16 rows x 64 cols = 4 accumulators; 8 WMMAs per K-tile.
// A tile is fetched by the Tensor Data Mover (padding done by TDM);
// B is transposed into LDS ([n][k]) so fragment reads are contiguous b128.
// Epilogue: +bias[n], + dscale[n]*res[m,n] (dscale==null -> 1.0), f32/f16 out.
// Requires: M%64==0, N%128==0, K%64==0.
// ---------------------------------------------------------------------------
__global__ __launch_bounds__(256) void gemm_wmma_f16(
    const _Float16* __restrict__ A, const _Float16* __restrict__ B,
    int N, int K,
    const float* __restrict__ bias,
    const float* __restrict__ res, const float* __restrict__ dscale,
    float* __restrict__ out32, _Float16* __restrict__ out16)
{
  __shared__ __align__(16) _Float16 As[64][72];    // [m][k], padded rows (144B)
  __shared__ __align__(16) _Float16 Bst[128][72];  // [n][k] transposed, padded

  const int tid  = threadIdx.x;
  const int lane = tid & 31;
  const int wave = tid >> 5;
  const int wm   = wave >> 1;   // 0..3 -> 16-row strip
  const int wn   = wave & 1;    // 0..1 -> 64-col strip
  const int bm   = blockIdx.y * 64;
  const int bn   = blockIdx.x * 128;

  v8f acc[4] = {};

  const int arow = tid >> 2;          // 0..63
  const int acol = (tid & 3) << 4;    // 0,16,32,48
  const _Float16* Ap = A + (size_t)(bm + arow) * K + acol;
#ifdef USE_TDM
  const unsigned ldsA = (unsigned)(size_t)(&As[0][0]);
#endif

  for (int k0 = 0; k0 < K; k0 += 64) {
    if (k0 + 64 < K) {                 // global_prefetch_b8 next K-tile
      __builtin_prefetch(Ap + 64, 0, 1);
      __builtin_prefetch(B + (size_t)(k0 + 64) * N + bn + ((tid & 15) << 3), 0, 1);
    }
#ifdef USE_TDM
    if (wave == 0)
      tdm_load_tile64x64_f16(A + (size_t)bm * K + k0, ldsA, K);
#else
    uint4 a0 = *(const uint4*)(Ap);
    uint4 a1 = *(const uint4*)(Ap + 8);
    *(uint4*)(&As[arow][acol])     = a0;
    *(uint4*)(&As[arow][acol + 8]) = a1;
#endif
    Ap += 64;

    // B tile 64x128, transposed into Bst: pairs of K-rows -> packed b32 stores
#pragma unroll
    for (int rep = 0; rep < 2; ++rep) {
      const int idx = tid + (rep << 8);
      const int nc  = idx & 15;        // n-chunk of 8
      const int kp  = idx >> 4;        // k-pair 0..31
      const _Float16* bp = B + (size_t)(k0 + 2 * kp) * N + bn + (nc << 3);
      v8h lo = *(const v8h*)(bp);
      v8h hi = *(const v8h*)(bp + N);
#pragma unroll
      for (int j = 0; j < 8; ++j) {
        h2 p; p[0] = lo[j]; p[1] = hi[j];
        *(h2*)(&Bst[(nc << 3) + j][2 * kp]) = p;
      }
    }
#ifdef USE_TDM
    if (wave == 0) __builtin_amdgcn_s_wait_tensorcnt(0);
#endif
    __syncthreads();

    const int m   = (wm << 4) + (lane & 15);
    const int kb  = (lane < 16) ? 0 : 8;   // A: lanes 0-15 K0-7/16-23, 16-31 K8-15/24-31
    const int kb2 = (lane < 16) ? 0 : 16;  // B: same K order as A
#pragma unroll
    for (int kh = 0; kh < 2; ++kh) {
      v16h af;
#pragma unroll
      for (int r = 0; r < 8; ++r) {
        const int kk = 32 * kh + ((r < 4) ? (2 * r) : (16 + 2 * (r - 4))) + kb;
        af[2 * r]     = As[m][kk];
        af[2 * r + 1] = As[m][kk + 1];
      }
#pragma unroll
      for (int s = 0; s < 4; ++s) {
        const int n = (wn << 6) + (s << 4) + (lane & 15);
        v16h bf;
#pragma unroll
        for (int e = 0; e < 16; ++e) bf[e] = Bst[n][32 * kh + kb + ((e < 8) ? e : (8 + e))];
        acc[s] = __builtin_amdgcn_wmma_f32_16x16x32_f16(false, af, false, bf,
                                                        (short)0, acc[s], false, false);
      }
    }
    __syncthreads();
  }

  // epilogue; D layout: VGPR v -> row v (lanes 0-15) / v+8 (lanes 16-31)
  const int rbase = bm + (wm << 4) + ((lane >> 4) << 3);
#pragma unroll
  for (int s = 0; s < 4; ++s) {
    const int col = bn + (wn << 6) + (s << 4) + (lane & 15);
    const float bv = bias ? bias[col] : 0.f;
    const float dv = dscale ? dscale[col] : 1.f;
#pragma unroll
    for (int v = 0; v < 8; ++v) {
      const size_t idx = (size_t)(rbase + v) * N + col;
      float val = acc[s][v] + bv;
      if (res)   val += dv * res[idx];
      if (out32) out32[idx] = val;
      if (out16) out16[idx] = (_Float16)val;
    }
  }
}

// ---------------------------------------------------------------------------
// Fused W1-GEMM + GLU: out16[m, n] = (A@W1 + b1)[m,n] * sigmoid((A@W1+b1)[m,n+512])
// for n in [0,512). Block computes col tiles [bn,bn+64) and [bn+512,bn+576)
// simultaneously; the [M,1024] intermediate never hits memory.
// B = W1 [K,1024] f16. Requires K%64==0.
// ---------------------------------------------------------------------------
__global__ __launch_bounds__(256) void gemm_wmma_glu(
    const _Float16* __restrict__ A, const _Float16* __restrict__ B,
    int K, const float* __restrict__ bias, _Float16* __restrict__ out16)
{
  __shared__ __align__(16) _Float16 As[64][72];
  __shared__ __align__(16) _Float16 Ba[64][72];   // cols bn..bn+63, [n][k]
  __shared__ __align__(16) _Float16 Bb[64][72];   // cols bn+512..bn+575, [n][k]

  const int tid  = threadIdx.x;
  const int lane = tid & 31;
  const int wave = tid >> 5;
  const int wm   = wave >> 1;   // 0..3
  const int wn   = wave & 1;    // 0..1 -> 32-col strip
  const int bm   = blockIdx.y * 64;
  const int bn   = blockIdx.x * 64;

  v8f accA[2] = {};
  v8f accB[2] = {};

  const int arow = tid >> 2;
  const int acol = (tid & 3) << 4;
  const _Float16* Ap = A + (size_t)(bm + arow) * K + acol;
#ifdef USE_TDM
  const unsigned ldsA = (unsigned)(size_t)(&As[0][0]);
#endif

  const int nc = tid & 7;    // n-chunk of 8 -> 64 cols
  const int kp = tid >> 3;   // k-pair 0..31

  for (int k0 = 0; k0 < K; k0 += 64) {
    if (k0 + 64 < K) __builtin_prefetch(Ap + 64, 0, 1);

#ifdef USE_TDM
    if (wave == 0)
      tdm_load_tile64x64_f16(A + (size_t)bm * K + k0, ldsA, K);
#else
    uint4 a0 = *(const uint4*)(Ap);
    uint4 a1 = *(const uint4*)(Ap + 8);
    *(uint4*)(&As[arow][acol])     = a0;
    *(uint4*)(&As[arow][acol + 8]) = a1;
#endif
    Ap += 64;

    const _Float16* bp = B + (size_t)(k0 + 2 * kp) * LDIM + bn + (nc << 3);
    v8h loA = *(const v8h*)(bp);
    v8h hiA = *(const v8h*)(bp + LDIM);
    v8h loB = *(const v8h*)(bp + 512);
    v8h hiB = *(const v8h*)(bp + 512 + LDIM);
#pragma unroll
    for (int j = 0; j < 8; ++j) {
      h2 pa; pa[0] = loA[j]; pa[1] = hiA[j];
      h2 pb; pb[0] = loB[j]; pb[1] = hiB[j];
      *(h2*)(&Ba[(nc << 3) + j][2 * kp]) = pa;
      *(h2*)(&Bb[(nc << 3) + j][2 * kp]) = pb;
    }
#ifdef USE_TDM
    if (wave == 0) __builtin_amdgcn_s_wait_tensorcnt(0);
#endif
    __syncthreads();

    const int m   = (wm << 4) + (lane & 15);
    const int kb  = (lane < 16) ? 0 : 8;
    const int kb2 = (lane < 16) ? 0 : 16;
#pragma unroll
    for (int kh = 0; kh < 2; ++kh) {
      v16h af;
#pragma unroll
      for (int r = 0; r < 8; ++r) {
        const int kk = 32 * kh + ((r < 4) ? (2 * r) : (16 + 2 * (r - 4))) + kb;
        af[2 * r]     = As[m][kk];
        af[2 * r + 1] = As[m][kk + 1];
      }
#pragma unroll
      for (int s = 0; s < 2; ++s) {
        const int n = (wn << 5) + (s << 4) + (lane & 15);
        v16h bfA, bfB;
#pragma unroll
        for (int e = 0; e < 16; ++e) bfA[e] = Ba[n][32 * kh + kb + ((e < 8) ? e : (8 + e))];
#pragma unroll
        for (int e = 0; e < 16; ++e) bfB[e] = Bb[n][32 * kh + kb + ((e < 8) ? e : (8 + e))];
        accA[s] = __builtin_amdgcn_wmma_f32_16x16x32_f16(false, af, false, bfA,
                                                         (short)0, accA[s], false, false);
        accB[s] = __builtin_amdgcn_wmma_f32_16x16x32_f16(false, af, false, bfB,
                                                         (short)0, accB[s], false, false);
      }
    }
    __syncthreads();
  }

  const int rbase = bm + (wm << 4) + ((lane >> 4) << 3);
#pragma unroll
  for (int s = 0; s < 2; ++s) {
    const int col = bn + (wn << 5) + (s << 4) + (lane & 15);
    const float ba = bias[col];
    const float bb = bias[col + 512];
#pragma unroll
    for (int v = 0; v < 8; ++v) {
      const float a = accA[s][v] + ba;
      const float g = accB[s][v] + bb;
      out16[(size_t)(rbase + v) * 512 + col] =
          (_Float16)(a * (1.f / (1.f + expf(-g))));
    }
  }
}

// ---------------------------------------------------------------------------
// small helper kernels
// ---------------------------------------------------------------------------
__global__ void zero_f32(float* p, int n) {
  int i = blockIdx.x * blockDim.x + threadIdx.x;
  if (i < n) p[i] = 0.f;
}

__global__ void cvt_f16(const float* __restrict__ in, _Float16* __restrict__ out, size_t n) {
  size_t j = (size_t)blockIdx.x * blockDim.x + threadIdx.x;
  if (j < n) out[j] = (_Float16)in[j];
}

__global__ void prep_lam(const float* __restrict__ nu, const float* __restrict__ th,
                         float* __restrict__ lre, float* __restrict__ lim,
                         float* __restrict__ gam, int n) {
  int i = blockIdx.x * blockDim.x + threadIdx.x;
  if (i >= n) return;
  float r = expf(-expf(nu[i]));
  float a = expf(th[i]);
  float re = r * cosf(a), im = r * sinf(a);
  lre[i] = re; lim[i] = im;
  gam[i] = sqrtf(fmaxf(1.f - (re * re + im * im), 1e-12f));
}

// Interleaved re/im columns: Bw[i][e][2h] = B_re[i][h][e]*g, [2h+1] = B_im*g
__global__ void prep_Bw(const float* __restrict__ Bre, const float* __restrict__ Bim,
                        const float* __restrict__ gam, _Float16* __restrict__ Bw, int n) {
  int idx = blockIdx.x * blockDim.x + threadIdx.x;
  if (idx >= n) return;
  int e = idx & 255;
  int h = (idx >> 8) & 255;
  int i = idx >> 16;
  float g = gam[i * HDIM + h];
  size_t src = ((size_t)i * HDIM + h) * EDIM + e;
  size_t dst = ((size_t)i * EDIM + e) * 512 + 2 * h;
  Bw[dst]     = (_Float16)(Bre[src] * g);
  Bw[dst + 1] = (_Float16)(Bim[src] * g);
}

// Interleaved re/im rows: Cw[i][2h][e] = C_re[i][e][h], [2h+1][e] = -C_im[i][e][h]
__global__ void prep_Cw(const float* __restrict__ Cre, const float* __restrict__ Cim,
                        _Float16* __restrict__ Cw, int n) {
  int idx = blockIdx.x * blockDim.x + threadIdx.x;
  if (idx >= n) return;
  int h = idx & 255;
  int e = (idx >> 8) & 255;
  int i = idx >> 16;
  size_t src = ((size_t)i * EDIM + e) * HDIM + h;
  Cw[((size_t)i * 512 + 2 * h) * EDIM + e]     = (_Float16)Cre[src];
  Cw[((size_t)i * 512 + 2 * h + 1) * EDIM + e] = (_Float16)(-Cim[src]);
}

__global__ void bn_partial(const float* __restrict__ x, float* __restrict__ sum,
                           float* __restrict__ sumsq, int rowsPerBlock) {
  int e = threadIdx.x;
  size_t r0 = (size_t)blockIdx.x * rowsPerBlock;
  float s = 0.f, s2 = 0.f;
  for (int m = 0; m < rowsPerBlock; ++m) {
    float v = x[(r0 + m) * EDIM + e];
    s += v; s2 += v * v;
  }
  atomicAdd(&sum[e], s);
  atomicAdd(&sumsq[e], s2);
}

__global__ void bn_finalize(float* stat, float invM) {
  int e = threadIdx.x;
  float mu  = stat[e] * invM;
  float var = stat[256 + e] * invM - mu * mu;
  stat[512 + e] = mu;
  stat[768 + e] = rsqrtf(var + 1e-5f);
}

__global__ void bn_apply(const float* __restrict__ x, const float* __restrict__ stat,
                         const float* __restrict__ scale, const float* __restrict__ bias,
                         float* __restrict__ h32, _Float16* __restrict__ h16, size_t n) {
  size_t j = (size_t)blockIdx.x * blockDim.x + threadIdx.x;
  if (j >= n) return;
  int e = (int)(j & 255);
  float v = (x[j] - stat[512 + e]) * stat[768 + e] * scale[e] + bias[e];
  h32[j] = v;
  h16[j] = (_Float16)v;
}

// complex diagonal recurrence; u/hs: [B*T, 256] of interleaved (re,im) f16 pairs
__global__ __launch_bounds__(256) void lru_scan(
    const h2* __restrict__ u, h2* __restrict__ hs,
    const float* __restrict__ lam_re, const float* __restrict__ lam_im) {
  int h = threadIdx.x;
  int b = blockIdx.x;
  float lr = lam_re[h], li = lam_im[h];
  float hr = 0.f, hi = 0.f;
  size_t base = (size_t)b * TDIM * 256 + h;
  for (int t = 0; t < TDIM; ++t) {
    size_t off = base + (size_t)t * 256;
    h2 uv = u[off];
    float nr = fmaf(lr, hr, fmaf(-li, hi, (float)uv[0]));
    float ni = fmaf(lr, hi, fmaf( li, hr, (float)uv[1]));
    hr = nr; hi = ni;
    h2 o; o[0] = (_Float16)hr; o[1] = (_Float16)hi;
    hs[off] = o;
  }
}

__global__ void pool_kernel(const float* __restrict__ x, float* __restrict__ pooled) {
  int e = threadIdx.x;
  int b = blockIdx.x;
  float s = 0.f;
  size_t base = (size_t)b * TDIM * EDIM;
  for (int t = 0; t < TDIM; ++t) s += x[base + (size_t)t * EDIM + e];
  pooled[b * EDIM + e] = s * (1.f / TDIM);
}

__global__ void head_kernel(const float* __restrict__ pooled, const float* __restrict__ W,
                            const float* __restrict__ bout, float* __restrict__ out, int n) {
  int i = blockIdx.x * blockDim.x + threadIdx.x;
  if (i >= n) return;
  int b = i / ODIM, o = i % ODIM;
  float s = bout[o];
  for (int e = 0; e < EDIM; ++e) s += pooled[b * EDIM + e] * W[e * ODIM + o];
  out[i] = s;
}

// ---------------------------------------------------------------------------
extern "C" void kernel_launch(void* const* d_in, const int* in_sizes, int n_in,
                              void* d_out, int out_size, void* d_ws, size_t ws_size,
                              hipStream_t stream) {
  const float* x_in   = (const float*)d_in[0];
  const float* W_enc  = (const float*)d_in[1];
  const float* b_enc  = (const float*)d_in[2];
  const float* nu_log = (const float*)d_in[3];
  const float* th_log = (const float*)d_in[4];
  const float* B_re   = (const float*)d_in[5];
  const float* B_im   = (const float*)d_in[6];
  const float* C_re   = (const float*)d_in[7];
  const float* C_im   = (const float*)d_in[8];
  const float* Dvec   = (const float*)d_in[9];
  const float* W1     = (const float*)d_in[10];
  const float* b1     = (const float*)d_in[11];
  const float* W2     = (const float*)d_in[12];
  const float* b2     = (const float*)d_in[13];
  const float* bn_sc  = (const float*)d_in[14];
  const float* bn_bi  = (const float*)d_in[15];
  const float* W_out  = (const float*)d_in[16];
  const float* b_out  = (const float*)d_in[17];
  float* out = (float*)d_out;

  const size_t M = (size_t)MROWS;
  char* ws = (char*)d_ws;
  size_t off = 0;
  auto carve = [&](size_t bytes) -> char* {
    char* p = ws + off;
    off += (bytes + 255) & ~(size_t)255;
    return p;
  };

  float*    y32   = (float*)carve(M * EDIM * 4);
  float*    x32   = (float*)carve(M * EDIM * 4);
  float*    h32   = (float*)carve(M * EDIM * 4);
  _Float16* a16A  = (_Float16*)carve(M * 512 * 2);
  _Float16* a16B  = (_Float16*)carve(M * 512 * 2);
  _Float16* u16   = (_Float16*)carve(M * 512 * 2);
  _Float16* x16   = (_Float16*)carve(M * TSDIM * 2);
  float*    pooled= (float*)carve(BDIM * EDIM * 4);
  float*    bnstat= (float*)carve(1024 * 4);
  float*    lamre = (float*)carve(NBLK * HDIM * 4);
  float*    lamim = (float*)carve(NBLK * HDIM * 4);
  float*    gam   = (float*)carve(NBLK * HDIM * 4);
  _Float16* enc16 = (_Float16*)carve((size_t)TSDIM * EDIM * 2);
  _Float16* Bw    = (_Float16*)carve((size_t)NBLK * EDIM * 512 * 2);
  _Float16* Cw    = (_Float16*)carve((size_t)NBLK * 512 * EDIM * 2);
  _Float16* W1w   = (_Float16*)carve((size_t)NBLK * EDIM * LDIM * 2);
  _Float16* W2w   = (_Float16*)carve((size_t)NBLK * 512 * EDIM * 2);
  (void)ws_size; (void)in_sizes; (void)n_in; (void)out_size;

  // ---- weight prep -------------------------------------------------------
  {
    int n = NBLK * HDIM;
    prep_lam<<<(n + 255) / 256, 256, 0, stream>>>(nu_log, th_log, lamre, lamim, gam, n);
  }
  {
    size_t n = (size_t)TSDIM * EDIM;
    cvt_f16<<<(int)((n + 255) / 256), 256, 0, stream>>>(W_enc, enc16, n);
  }
  {
    size_t n = (size_t)NBLK * EDIM * LDIM;
    cvt_f16<<<(int)((n + 255) / 256), 256, 0, stream>>>(W1, W1w, n);
  }
  {
    size_t n = (size_t)NBLK * 512 * EDIM;
    cvt_f16<<<(int)((n + 255) / 256), 256, 0, stream>>>(W2, W2w, n);
  }
  {
    size_t n = M * TSDIM;
    cvt_f16<<<(int)((n + 255) / 256), 256, 0, stream>>>(x_in, x16, n);
  }
  {
    int n = NBLK * HDIM * EDIM;
    prep_Bw<<<(n + 255) / 256, 256, 0, stream>>>(B_re, B_im, gam, Bw, n);
    prep_Cw<<<(n + 255) / 256, 256, 0, stream>>>(C_re, C_im, Cw, n);
  }

  // ---- encoder: y = x @ W_enc + b_enc ------------------------------------
  {
    dim3 grid(EDIM / 128, (unsigned)(M / 64));
    gemm_wmma_f16<<<grid, 256, 0, stream>>>(x16, enc16, EDIM, TSDIM,
                                            b_enc, nullptr, nullptr, y32, nullptr);
  }

  // ---- 6 residual LRU blocks ---------------------------------------------
  for (int i = 0; i < NBLK; ++i) {
    const float* xin = (i == 0) ? y32 : x32;

    // batchnorm over (B,T)
    zero_f32<<<2, 256, 0, stream>>>(bnstat, 512);
    bn_partial<<<64, 256, 0, stream>>>(xin, bnstat, bnstat + 256, (int)(M / 64));
    bn_finalize<<<1, 256, 0, stream>>>(bnstat, 1.f / (float)M);
    bn_apply<<<(int)((M * EDIM + 255) / 256), 256, 0, stream>>>(
        xin, bnstat, bn_sc + i * EDIM, bn_bi + i * EDIM, h32, a16A, M * EDIM);

    // Bu = h @ Bw_i (re/im interleaved cols, gamma folded) -> u16 [M,512]
    {
      dim3 grid(512 / 128, (unsigned)(M / 64));
      gemm_wmma_f16<<<grid, 256, 0, stream>>>(a16A, Bw + (size_t)i * EDIM * 512,
                                              512, EDIM, nullptr, nullptr, nullptr,
                                              nullptr, u16);
    }

    // sequential complex scan -> hs16 (interleaved) in a16B
    lru_scan<<<BDIM, 256, 0, stream>>>((const h2*)u16, (h2*)a16B,
                                       lamre + i * HDIM, lamim + i * HDIM);

    // lru_out = hs @ Cw_i + D*h   -> a16A [M,256]
    {
      dim3 grid(EDIM / 128, (unsigned)(M / 64));
      gemm_wmma_f16<<<grid, 256, 0, stream>>>(a16B, Cw + (size_t)i * 512 * EDIM,
                                              EDIM, 512, nullptr, h32, Dvec + i * EDIM,
                                              nullptr, a16A);
    }

    // fused: glu = GLU(lru_out @ W1 + b1)  -> a16B [M,512]
    {
      dim3 grid(512 / 64, (unsigned)(M / 64));
      gemm_wmma_glu<<<grid, 256, 0, stream>>>(a16A, W1w + (size_t)i * EDIM * LDIM,
                                              EDIM, b1 + i * LDIM, a16B);
    }

    // x = glu @ W2 + b2 + y  -> x32 [M,256]
    {
      dim3 grid(EDIM / 128, (unsigned)(M / 64));
      gemm_wmma_f16<<<grid, 256, 0, stream>>>(a16B, W2w + (size_t)i * 512 * EDIM,
                                              EDIM, 512, b2 + i * EDIM, y32, nullptr,
                                              x32, nullptr);
    }
  }

  // ---- mean-pool over T + head -------------------------------------------
  pool_kernel<<<BDIM, 256, 0, stream>>>(x32, pooled);
  head_kernel<<<(BDIM * ODIM + 255) / 256, 256, 0, stream>>>(pooled, W_out, b_out, out,
                                                             BDIM * ODIM);
}
